// GraphAttentionLayer_75118978007306
// MI455X (gfx1250) — matrix operand order repaired and checked by reading, not compiled, not run on hardware
//
#include <hip/hip_runtime.h>
#include <hip/hip_bf16.h>
#include <math.h>

#define B_ 2
#define N_ 2048
#define D_ 256
#define H_ 8
#define HD_ 32
#define ROWS (B_*N_)   // 4096
#define APAD 264       // 256 + 8 f16 pad -> conflict-free ds_load_b128

typedef _Float16 v16h __attribute__((ext_vector_type(16)));
typedef _Float16 v8h  __attribute__((ext_vector_type(8)));
typedef float    v8f  __attribute__((ext_vector_type(8)));
typedef int      i4v  __attribute__((vector_size(16)));   // matches builtin's int4* param

#if defined(__has_builtin)
# if __has_builtin(__builtin_amdgcn_global_load_async_to_lds_b128)
#  define HAVE_ASYNC_B128 1
# endif
# if __has_builtin(__builtin_amdgcn_s_wait_asynccnt)
#  define HAVE_WAIT_ASYNC 1
# endif
#endif

#if defined(HAVE_ASYNC_B128) && defined(HAVE_WAIT_ASYNC)
#define USE_ASYNC_LDS 1
#endif

// 16-byte global -> LDS copy. Async (ASYNCcnt) when the toolchain has the
// gfx1250 builtins, else load+ds_store fallback.
__device__ __forceinline__ void copy16_g2l(const _Float16* g, _Float16* l){
#if defined(USE_ASYNC_LDS)
  __builtin_amdgcn_global_load_async_to_lds_b128((i4v*)g, (i4v*)l, 0, 0);
#else
  *(v8h*)l = *(const v8h*)g;
#endif
}
__device__ __forceinline__ void copy_fence(){
#if defined(USE_ASYNC_LDS)
  __builtin_amdgcn_s_wait_asynccnt(0);
#endif
}

__device__ __forceinline__ v8f wmma_f16(v16h a, v16h b, v8f c){
  return __builtin_amdgcn_wmma_f32_16x16x32_f16(false, a, false, b, (short)0, c, false, false);
}

// A fragment (16x32 f16): K(e) = e + 16*half, row = lane&15.
__device__ __forceinline__ v16h load_afrag(const _Float16* p, int half){
  v8h lo = *(const v8h*)(p + 16*half);
  v8h hi = *(const v8h*)(p + 16*half + 8);
  v16h r;
  #pragma unroll
  for (int i=0;i<8;i++){ r[i]=lo[i]; r[8+i]=hi[i]; }
  return r;
}
// B fragment (32x16 f16) from K-contiguous storage: K = i + 16*half, col = lane&15.
__device__ __forceinline__ v16h load_bfrag(const _Float16* p, int half){
  return *(const v16h*)(p + 16*half);
}

// ---------------- prep kernels ----------------
__global__ void k_cvt_f16(const float* __restrict__ src, _Float16* __restrict__ dst, int n){
  int i = blockIdx.x*blockDim.x + threadIdx.x;
  if (i < n) dst[i] = (_Float16)src[i];
}

__global__ void k_cvt_wt(const float* __restrict__ W, _Float16* __restrict__ WT){
  int t = blockIdx.x*blockDim.x + threadIdx.x; // 65536
  int n = t >> 8, k = t & 255;
  WT[n*D_ + k] = (_Float16)W[k*D_ + n];
}

// fold attn_a into Wq/Wk: uqT[h*256+k] = sum_e Wq[k, h*32+e]*a_q[h,e]
__global__ void k_fold_attn(const float* __restrict__ Wq, const float* __restrict__ Wk,
                            const float* __restrict__ attn_a,
                            float* __restrict__ uqT, float* __restrict__ ukT){
  int t = blockIdx.x*blockDim.x + threadIdx.x; // 4096
  int isk = t >> 11; int h = (t>>8)&7; int k = t&255;
  const float* W = isk ? Wk : Wq;
  const float* a = attn_a + h*2*HD_ + isk*HD_;
  float s = 0.f;
  #pragma unroll
  for (int e=0;e<HD_;e++) s += W[k*D_ + h*HD_ + e]*a[e];
  (isk?ukT:uqT)[h*D_ + k] = s;
}

// sq[b,n,h] = x[row]·uq[:,h] + bias-part ; head-major sqT[h*4096 + row]
__global__ void k_sqsk(const float* __restrict__ x,
                       const float* __restrict__ uqT, const float* __restrict__ ukT,
                       const float* __restrict__ bq, const float* __restrict__ bk,
                       const float* __restrict__ attn_a,
                       float* __restrict__ sqT, float* __restrict__ skT){
  int t = blockIdx.x*blockDim.x + threadIdx.x; // 65536
  int g = t & (ROWS-1); int h = (t>>12)&7; int isk = t>>15;
  const float* u  = (isk?ukT:uqT) + h*D_;
  const float* xr = x + (size_t)g*D_;
  float s = 0.f;
  for (int k=0;k<D_;k++) s += xr[k]*u[k];
  const float* bias = isk?bk:bq;
  const float* a = attn_a + h*2*HD_ + isk*HD_;
  float sb = 0.f;
  #pragma unroll
  for (int e=0;e<HD_;e++) sb += bias[h*HD_+e]*a[e];
  (isk?skT:sqT)[h*ROWS + g] = s + sb;
}

// cooperative A-tile stage: 16 rows x 256 f16 -> LDS (padded rows)
__device__ __forceinline__ void stage_atile(const _Float16* __restrict__ A, int rt,
                                            _Float16* sA){
  #pragma unroll
  for (int c = 0; c < 2; c++){
    int idx = threadIdx.x + c*256;        // 512 chunks of 16B
    int row = idx >> 5, col = (idx & 31)*8;
    copy16_g2l(A + (size_t)(rt*16 + row)*D_ + col, sA + row*APAD + col);
  }
  copy_fence();
  __syncthreads();
}

// ---------------- V = x @ Wv + bv (WMMA), stored transposed f16 vT[d][row] ----------------
__global__ void k_gemm_v(const _Float16* __restrict__ x16, const _Float16* __restrict__ WvT,
                         const float* __restrict__ bv, _Float16* __restrict__ vT){
  __shared__ __align__(16) _Float16 sA[16*APAD];
  int rt = blockIdx.x >> 1, cb = (blockIdx.x & 1)*8;   // 256 row tiles x 2 col groups
  stage_atile(x16, rt, sA);

  int wave = threadIdx.x >> 5, lane = threadIdx.x & 31;
  int ct = cb + wave;
  int m = lane & 15, half = lane >> 4;
  int ng = ct*16 + m;                                  // output column (d)
  v8f acc = {};
  const _Float16* arow = sA + m*APAD;
  const _Float16* bcol = WvT + (size_t)ng*D_;
  #pragma unroll
  for (int kt=0; kt<D_; kt+=32)
    acc = wmma_f16(load_afrag(arow + kt, half), load_bfrag(bcol + kt, half), acc);
  float bias = bv[ng];
  v8h outv;
  #pragma unroll
  for (int r=0;r<8;r++) outv[r] = (_Float16)(acc[r] + bias);
  *(v8h*)(vT + (size_t)ng*ROWS + rt*16 + 8*half) = outv;  // C rows contiguous -> b128
}

// ---------------- fused graph attention ----------------
__global__ void k_attn(const int* __restrict__ adj,
                       const float* __restrict__ sqT, const float* __restrict__ skT,
                       const _Float16* __restrict__ vT,
                       _Float16* __restrict__ att16, float* __restrict__ mw_out){
  __shared__ float s_r[256];
  __shared__ float st_m[128], st_rz[128];
  __shared__ float mw[512];              // 16x32 head-summed weight tile
  int t = threadIdx.x;
  int b  = blockIdx.x >> 7;
  int it = blockIdx.x & 127;

  // ---- pass 1: softmax stats per (row, head); 2 threads per pair ----
  {
    int p = t >> 1, s = t & 1;
    int il = p >> 3, h = p & 7;
    int ig = it*16 + il;
    float sqv = sqT[h*ROWS + b*N_ + ig];
    const int*   arow  = adj + (size_t)(b*N_ + ig)*N_;
    const float* skrow = skT + h*ROWS + b*N_;
    // phase A: masked max of sk (LeakyReLU is monotonic -> fold through later)
    float mk = -1e30f;
    for (int j = s*4; j < N_; j += 8){
      int4   a4 = *(const int4*)(arow + j);
      float4 k4 = *(const float4*)(skrow + j);
      if (a4.x) mk = fmaxf(mk, k4.x);
      if (a4.y) mk = fmaxf(mk, k4.y);
      if (a4.z) mk = fmaxf(mk, k4.z);
      if (a4.w) mk = fmaxf(mk, k4.w);
    }
    s_r[t] = mk; __syncthreads();
    if (s == 0){
      float mm = fmaxf(s_r[t], s_r[t+1]);
      float sc = sqv + mm;
      st_m[p] = sc >= 0.f ? sc : 0.2f*sc;
    }
    __syncthreads();
    // phase B: sum of exp (independent exps -> good TRANS-pipe ILP)
    float m = st_m[p];
    float Z = 0.f;
    for (int j = s*4; j < N_; j += 8){
      int4   a4 = *(const int4*)(arow + j);
      float4 k4 = *(const float4*)(skrow + j);
      float sc;
      sc = sqv + k4.x; sc = sc>=0.f?sc:0.2f*sc; Z += a4.x ? __expf(sc-m) : 0.f;
      sc = sqv + k4.y; sc = sc>=0.f?sc:0.2f*sc; Z += a4.y ? __expf(sc-m) : 0.f;
      sc = sqv + k4.z; sc = sc>=0.f?sc:0.2f*sc; Z += a4.z ? __expf(sc-m) : 0.f;
      sc = sqv + k4.w; sc = sc>=0.f?sc:0.2f*sc; Z += a4.w ? __expf(sc-m) : 0.f;
    }
    s_r[t] = Z; __syncthreads();
    if (s == 0){
      float Z2 = s_r[t] + s_r[t+1];
      st_rz[p] = Z2 > 0.f ? 1.f/Z2 : 0.f;
    }
    __syncthreads();
  }

  // ---- pass 2: wave h = head h; w-fragment in registers, 2 WMMAs per j-tile ----
  int wave = t >> 5, lane = t & 31;
  int h = wave;
  int m_l = lane & 15, half = lane >> 4;
  int ig = it*16 + m_l;
  float sqv = sqT[h*ROWS + b*N_ + ig];
  float ms  = st_m[m_l*8 + h];
  float rz  = st_rz[m_l*8 + h];
  const int*   arow  = adj + (size_t)(b*N_ + ig)*N_;
  const float* skrow = skT + h*ROWS + b*N_;
  v8f acc0 = {}, acc1 = {};
  float* mwrow = mw_out + (size_t)(b*N_ + it*16)*N_;

  for (int jt = 0; jt < N_; jt += 32){
    mw[t] = 0.f; mw[t+256] = 0.f;
    __syncthreads();
    __builtin_prefetch(arow + jt + 64, 0, 1);          // global_prefetch_b8
    // vectorized tile loads: e 0..15 -> j = jt+16*half+e
    int b0i = jt + 16*half, b1i = jt + 16*half + 8;
    int4   a0 = *(const int4*)(arow + b0i), a1 = *(const int4*)(arow + b0i + 4);
    int4   a2 = *(const int4*)(arow + b1i), a3 = *(const int4*)(arow + b1i + 4);
    float4 k0 = *(const float4*)(skrow + b0i), k1 = *(const float4*)(skrow + b0i + 4);
    float4 k2 = *(const float4*)(skrow + b1i), k3 = *(const float4*)(skrow + b1i + 4);
    int   aa[16] = {a0.x,a0.y,a0.z,a0.w, a1.x,a1.y,a1.z,a1.w,
                    a2.x,a2.y,a2.z,a2.w, a3.x,a3.y,a3.z,a3.w};
    float kk[16] = {k0.x,k0.y,k0.z,k0.w, k1.x,k1.y,k1.z,k1.w,
                    k2.x,k2.y,k2.z,k2.w, k3.x,k3.y,k3.z,k3.w};
    v16h wfrag;
    #pragma unroll
    for (int e=0;e<16;e++){
      int kidx = e + 16*half;                          // A-fragment K mapping
      float we = 0.f;
      if (aa[e]){
        float sc = sqv + kk[e];
        sc = sc >= 0.f ? sc : 0.2f*sc;
        we = __expf(sc - ms)*rz;
      }
      wfrag[e] = (_Float16)we;
      atomicAdd(&mw[m_l*32 + kidx], we);               // ds_add_f32: sum over heads
    }
    v16h bf0 = *(const v16h*)(vT + (size_t)(h*HD_      + m_l)*ROWS + b*N_ + jt + 16*half);
    v16h bf1 = *(const v16h*)(vT + (size_t)(h*HD_ + 16 + m_l)*ROWS + b*N_ + jt + 16*half);
    acc0 = wmma_f16(wfrag, bf0, acc0);
    acc1 = wmma_f16(wfrag, bf1, acc1);
    __syncthreads();
    mwrow[((t      )>>5)*N_ + jt + (t&31)]       = mw[t]     * 0.125f;
    mwrow[((t + 256)>>5)*N_ + jt + ((t+256)&31)] = mw[t+256] * 0.125f;
  }

  #pragma unroll
  for (int r=0;r<8;r++){
    int row = b*N_ + it*16 + r + 8*half;               // C layout: M = r + 8*half
    att16[(size_t)row*D_ + h*HD_      + m_l] = (_Float16)acc0[r];
    att16[(size_t)row*D_ + h*HD_ + 16 + m_l] = (_Float16)acc1[r];
  }
}

// ---------------- out_pre = attended @ Wo + bo + x (WMMA, residual fused) ----------------
__global__ void k_gemm_o(const _Float16* __restrict__ a16, const _Float16* __restrict__ WoT,
                         const float* __restrict__ bo, const float* __restrict__ x,
                         float* __restrict__ outpre){
  __shared__ __align__(16) _Float16 sA[16*APAD];
  int rt = blockIdx.x >> 1, cb = (blockIdx.x & 1)*8;
  stage_atile(a16, rt, sA);

  int wave = threadIdx.x >> 5, lane = threadIdx.x & 31;
  int ct = cb + wave;
  int m = lane & 15, half = lane >> 4;
  int ng = ct*16 + m;
  v8f acc = {};
  const _Float16* arow = sA + m*APAD;
  const _Float16* bcol = WoT + (size_t)ng*D_;
  #pragma unroll
  for (int kt=0; kt<D_; kt+=32)
    acc = wmma_f16(load_afrag(arow + kt, half), load_bfrag(bcol + kt, half), acc);
  float bias = bo[ng];
  #pragma unroll
  for (int r=0;r<8;r++){
    int row = rt*16 + r + 8*half;
    outpre[(size_t)row*D_ + ng] = acc[r] + bias + x[(size_t)row*D_ + ng];
  }
}

// ---------------- LayerNorm: one wave32 per row ----------------
__global__ void k_ln(const float* __restrict__ op, const float* __restrict__ g,
                     const float* __restrict__ be, float* __restrict__ out){
  int wave = threadIdx.x >> 5, lane = threadIdx.x & 31;
  int row = blockIdx.x*8 + wave;
  const float* r = op + (size_t)row*D_;
  float v[8];
  float s = 0.f;
  #pragma unroll
  for (int i=0;i<8;i++){ v[i] = r[lane + 32*i]; s += v[i]; }
  #pragma unroll
  for (int o=16;o>0;o>>=1) s += __shfl_xor(s, o, 32);
  float mu = s * (1.f/D_);
  float vs = 0.f;
  #pragma unroll
  for (int i=0;i<8;i++){ float d = v[i]-mu; vs += d*d; }
  #pragma unroll
  for (int o=16;o>0;o>>=1) vs += __shfl_xor(vs, o, 32);
  float rs = rsqrtf(vs*(1.f/D_) + 1e-5f);
  #pragma unroll
  for (int i=0;i<8;i++){
    int c = lane + 32*i;
    out[(size_t)row*D_ + c] = (v[i]-mu)*rs*g[c] + be[c];
  }
}

extern "C" void kernel_launch(void* const* d_in, const int* in_sizes, int n_in,
                              void* d_out, int out_size, void* d_ws, size_t ws_size,
                              hipStream_t stream){
  (void)in_sizes; (void)n_in; (void)out_size; (void)ws_size;
  const float* x      = (const float*)d_in[0];
  const int*   adj    = (const int*)d_in[1];
  const float* Wq     = (const float*)d_in[2];
  const float* bq     = (const float*)d_in[3];
  const float* Wk     = (const float*)d_in[4];
  const float* bk     = (const float*)d_in[5];
  const float* Wv     = (const float*)d_in[6];
  const float* bv     = (const float*)d_in[7];
  const float* attn_a = (const float*)d_in[8];
  const float* Wo     = (const float*)d_in[9];
  const float* bo     = (const float*)d_in[10];
  const float* ln_g   = (const float*)d_in[11];
  const float* ln_b   = (const float*)d_in[12];
  float* out    = (float*)d_out;
  float* mw_out = out + (size_t)B_*N_*D_;

  char* ws = (char*)d_ws;
  _Float16* x16   = (_Float16*)(ws + 0);                        // 2 MB
  _Float16* vT    = (_Float16*)(ws + (2u<<20));                 // 2 MB [d][row]
  _Float16* att16 = (_Float16*)(ws + (4u<<20));                 // 2 MB
  _Float16* WvT   = (_Float16*)(ws + (6u<<20));                 // 128 KB
  _Float16* WoT   = (_Float16*)(ws + (6u<<20) + (1u<<17));      // 128 KB
  float*    uqT   = (float*)(ws + (6u<<20) + (2u<<17));         // 8 KB
  float*    ukT   = uqT + H_*D_;                                // 8 KB
  float*    sqT   = (float*)(ws + (7u<<20));                    // 128 KB
  float*    skT   = sqT + H_*ROWS;                              // 128 KB
  float*    outpre= (float*)(ws + (8u<<20));                    // 4 MB

  k_cvt_f16 <<<ROWS*D_/256, 256, 0, stream>>>(x, x16, ROWS*D_);
  k_cvt_wt  <<<256, 256, 0, stream>>>(Wv, WvT);
  k_cvt_wt  <<<256, 256, 0, stream>>>(Wo, WoT);
  k_fold_attn<<<16, 256, 0, stream>>>(Wq, Wk, attn_a, uqT, ukT);
  k_sqsk    <<<256, 256, 0, stream>>>(x, uqT, ukT, bq, bk, attn_a, sqT, skT);
  k_gemm_v  <<<512, 256, 0, stream>>>(x16, WvT, bv, vT);
  k_attn    <<<256, 256, 0, stream>>>(adj, sqT, skT, vT, att16, mw_out);
  k_gemm_o  <<<512, 256, 0, stream>>>(att16, WoT, bo, x, outpre);
  k_ln      <<<512, 256, 0, stream>>>(outpre, ln_g, ln_b, out);
}
